// CustomAttentionLayer_63385127354710
// MI455X (gfx1250) — matrix operand order repaired and checked by reading, not compiled, not run on hardware
//
#include <hip/hip_runtime.h>
#include <hip/hip_bf16.h>
#include <math.h>

// ---- problem constants (match reference) ----
#define ATT_W   64
#define SEQ     2048
#define DMODEL  512
#define BATCH   4
#define FCAT    1024   // concat feature dim
#define DOUT    512

typedef __attribute__((ext_vector_type(16))) _Float16 v16h;
typedef __attribute__((ext_vector_type(8)))  _Float16 v8h;
typedef __attribute__((ext_vector_type(4)))  _Float16 v4h;
typedef __attribute__((ext_vector_type(8)))  float    v8f;

union U16h { v16h v; v8h h[2]; };

__device__ __forceinline__ v8f zero8() {
  v8f z = {0.f, 0.f, 0.f, 0.f, 0.f, 0.f, 0.f, 0.f};
  return z;
}

// D = A*B + C, f16 inputs, f32 accumulate  ->  v_wmma_f32_16x16x32_f16
__device__ __forceinline__ v8f wmma_f16(v16h a, v16h b, v8f c) {
  return __builtin_amdgcn_wmma_f32_16x16x32_f16(
      /*neg_a=*/false, a, /*neg_b=*/false, b,
      /*c_mod=*/(short)0, c, /*reuse_a=*/false, /*reuse_b=*/false);
}

// A-matrix fragment (16xK, row-major f16 in LDS), K-chunk at kbase.
// Contiguous-k layout used here (same as B): lane l (m = l&15, half = l>>4):
//   halves[0..15] = A[m, kbase + half*16 + i]
__device__ __forceinline__ v16h lds_a_frag(const _Float16* buf, int rowStride, int kbase) {
  const int l    = threadIdx.x & 31;
  const int m    = l & 15;
  const int half = l >> 4;
  const _Float16* p = buf + m * rowStride + kbase + half * 16;
  U16h u;
  u.h[0] = *(const v8h*)(p);
  u.h[1] = *(const v8h*)(p + 8);
  return u.v;
}

// B-matrix fragment where B[k,n] = buf[n*rowStride + k] (B stored transposed,
// row-major f16 in global). Lane l: n = nbase + (l&15), 16 contiguous k's.
__device__ __forceinline__ v16h g_bt_frag_f16(const _Float16* buf, int rowStride,
                                              int nbase, int kbase) {
  const int l    = threadIdx.x & 31;
  const int n    = nbase + (l & 15);
  const int koff = kbase + (l >> 4) * 16;
  const _Float16* p = buf + (size_t)n * rowStride + koff;
  U16h u;
  u.h[0] = *(const v8h*)(p);
  u.h[1] = *(const v8h*)(p + 8);
  return u.v;
}

// B-fragment for scores: B[k, jr] = x[j(jr), k] read from global f32 with
// zero-fill for out-of-range rows (reproduces zero-padded window -> score 0).
// Returned as a union so the caller can also spill it to the LDS window tile.
__device__ __forceinline__ U16h b_frag_xrow(const float* __restrict__ xb,
                                            int j, int koff) {
  U16h u;
  if (j >= 0 && j < SEQ) {
    const float* p = xb + (size_t)j * DMODEL + koff;
#pragma unroll
    for (int i = 0; i < 16; i += 4) {
      float4 f = *(const float4*)(p + i);
      u.v[i + 0] = (_Float16)f.x;
      u.v[i + 1] = (_Float16)f.y;
      u.v[i + 2] = (_Float16)f.z;
      u.v[i + 3] = (_Float16)f.w;
    }
  } else {
#pragma unroll
    for (int i = 0; i < 16; ++i) u.v[i] = (_Float16)0.f;
  }
  return u;
}

// Pre-transpose W_c (1024x512 f32) -> whT (512x1024 f16), whT[o][f] = W_c[f][o]
__global__ void prep_wct_kernel(const float* __restrict__ Wc,
                                _Float16* __restrict__ whT) {
  int idx = blockIdx.x * blockDim.x + threadIdx.x;  // idx = o*1024 + f
  if (idx < DOUT * FCAT) {
    int o = idx >> 10;
    int f = idx & (FCAT - 1);
    whT[idx] = (_Float16)Wc[(size_t)f * DOUT + o];
  }
}

__global__ __launch_bounds__(128)
void attn_fused_kernel(const float* __restrict__ x,
                       const _Float16* __restrict__ whT,
                       float* __restrict__ out) {
  __shared__ _Float16 Ax[16 * DMODEL];   // x tile (rows t0..t0+15), f16 (16 KB)
  __shared__ _Float16 Xw[80 * DMODEL];   // window rows jr=0..79, f16 (80 KB)
  __shared__ float    Sc[16 * 80];       // raw scores (5 KB)
  __shared__ _Float16 Al[16 * 96];       // alpha, zero-padded band (3 KB)
  __shared__ _Float16 Cc[16 * DMODEL];   // context, f16 (16 KB)

  const int t0   = blockIdx.x * 16;
  const int b    = blockIdx.y;
  const int tid  = threadIdx.x;
  const int lane = tid & 31;
  const int wave = tid >> 5;
  const float* xb = x + (size_t)b * SEQ * DMODEL;

  // ---- Phase 0: stage 16x512 x tile as f16 in LDS ----
  for (int idx = tid * 4; idx < 16 * DMODEL; idx += 128 * 4) {
    const int m = idx >> 9;             // / DMODEL
    const int c = idx & (DMODEL - 1);
    float4 f = *(const float4*)(xb + (size_t)(t0 + m) * DMODEL + c);
    v4h h;
    h[0] = (_Float16)f.x; h[1] = (_Float16)f.y;
    h[2] = (_Float16)f.z; h[3] = (_Float16)f.w;
    *(v4h*)(Ax + idx) = h;
  }
  __syncthreads();

  // ---- Phase 1: scores S(16x80) = Xtile(16x512) @ Xwin^T(512x80).
  // Side effect: every converted window fragment is spilled to Xw so the
  // context GEMM (phase 3) can read f16 from LDS instead of strided global.
  for (int jt = wave; jt < 5; jt += 4) {         // wave-uniform tiles
    v8f acc = zero8();
    const int jrow  = jt * 16 + (lane & 15);
    const int j     = t0 - ATT_W + jrow;
    const int klane = (lane >> 4) * 16;
#pragma unroll 4
    for (int kc = 0; kc < DMODEL; kc += 32) {
      v16h a  = lds_a_frag(Ax, DMODEL, kc);
      U16h bu = b_frag_xrow(xb, j, kc + klane);
      // spill converted window data (zeros for j<0) into the LDS window tile
      _Float16* wp = Xw + jrow * DMODEL + kc + klane;
      *(v8h*)(wp)     = bu.h[0];
      *(v8h*)(wp + 8) = bu.h[1];
      acc = wmma_f16(a, bu.v, acc);
    }
    const int M0 = (lane >> 4) * 8;
#pragma unroll
    for (int r = 0; r < 8; ++r)
      Sc[(M0 + r) * 80 + jt * 16 + (lane & 15)] = acc[r];
  }
  __syncthreads();

  // ---- Phase 2: softmax over 64-wide band per row (padding scores == 0) ----
  if (tid < 16) {
    const int tt = tid;
    for (int k = 0; k < 96; ++k) Al[tt * 96 + k] = (_Float16)0.f;
    const float* srow = Sc + tt * 80;
    float mx = srow[tt];
    for (int w = 1; w < ATT_W; ++w) mx = fmaxf(mx, srow[tt + w]);
    float s = 0.f;
    for (int w = 0; w < ATT_W; ++w) s += __expf(srow[tt + w] - mx);
    const float inv = 1.f / s;
    for (int w = 0; w < ATT_W; ++w)
      Al[tt * 96 + tt + w] = (_Float16)(__expf(srow[tt + w] - mx) * inv);
  }
  __syncthreads();

  // ---- Phase 3: context C(16x512) = alpha(16x96) @ Xwin(96x512).
  // B-fragments gathered from the LDS window tile; rows 80..95 are K-padding
  // (alpha == 0 there) and read as zeros.
  for (int nt = wave; nt < 32; nt += 4) {
    v8f acc = zero8();
#pragma unroll
    for (int kc = 0; kc < 96; kc += 32) {
      v16h a = lds_a_frag(Al, 96, kc);
      const int d    = nt * 16 + (lane & 15);
      const int koff = kc + (lane >> 4) * 16;
      U16h bu;
#pragma unroll
      for (int i = 0; i < 16; ++i) {
        const int jr = koff + i;
        bu.v[i] = (jr < 80) ? Xw[jr * DMODEL + d] : (_Float16)0.f;
      }
      acc = wmma_f16(a, bu.v, acc);
    }
    const int M0 = (lane >> 4) * 8;
    const int d  = nt * 16 + (lane & 15);
#pragma unroll
    for (int r = 0; r < 8; ++r)
      Cc[(M0 + r) * DMODEL + d] = (_Float16)acc[r];
  }
  __syncthreads();

  // ---- Phase 4: out = sigmoid(concat([C, Xtile]) @ W_c), K = 1024 ----
  float* orow = out + ((size_t)b * SEQ + t0) * DMODEL;
  for (int nt = wave; nt < 32; nt += 4) {
    v8f acc = zero8();
    const int nrow = nt * 16 + (lane & 15);
#pragma unroll 4
    for (int kc = 0; kc < FCAT; kc += 32) {
      // speculative prefetch of the next W_c panel (global_prefetch_b8)
      if (kc + 128 < FCAT)
        __builtin_prefetch(whT + (size_t)nrow * FCAT + kc + 128, 0, 1);
      v16h a;
      if (kc < DMODEL) a = lds_a_frag(Cc, DMODEL, kc);
      else             a = lds_a_frag(Ax, DMODEL, kc - DMODEL);
      v16h bf = g_bt_frag_f16(whT, FCAT, nt * 16, kc);
      acc = wmma_f16(a, bf, acc);
    }
    const int M0 = (lane >> 4) * 8;
    const int o  = nt * 16 + (lane & 15);
#pragma unroll
    for (int r = 0; r < 8; ++r) {
      float v = acc[r];
      orow[(size_t)(M0 + r) * DMODEL + o] = 1.f / (1.f + __expf(-v));
    }
  }
}

extern "C" void kernel_launch(void* const* d_in, const int* in_sizes, int n_in,
                              void* d_out, int out_size, void* d_ws, size_t ws_size,
                              hipStream_t stream) {
  const float* x  = (const float*)d_in[0];   // (4, 2048, 512) f32
  const float* Wc = (const float*)d_in[1];   // (1024, 512) f32
  float* out      = (float*)d_out;           // (4, 2048, 512) f32
  _Float16* whT   = (_Float16*)d_ws;         // 512*1024 f16 = 1 MB scratch

  prep_wct_kernel<<<(DOUT * FCAT + 255) / 256, 256, 0, stream>>>(Wc, whT);

  dim3 grid(SEQ / 16, BATCH);
  attn_fused_kernel<<<grid, 128, 0, stream>>>(x, whT, out);
}
